// WindowAttention_66829691126267
// MI455X (gfx1250) — matrix operand order repaired and checked by reading, not compiled, not run on hardware
//
#include <hip/hip_runtime.h>
#include <hip/hip_bf16.h>

typedef __attribute__((ext_vector_type(16))) _Float16 v16h;
typedef __attribute__((ext_vector_type(8)))  _Float16 v8h;
typedef __attribute__((ext_vector_type(8)))  float    v8f;

#define DIM 192
#define H 6
#define HD 32
#define NTOK 49
#define NPAD 64
#define BWIN 2048
#define MROWS (BWIN * NTOK)        // 100352 -> 6272 M-tiles of 16
#define SCALE 0.17677669529663689f // 1/sqrt(32)

// merge two aligned 8-half runs into a WMMA A/B operand register
__device__ __forceinline__ v16h merge16(v8h lo, v8h hi) {
    return __builtin_shufflevector(lo, hi, 0, 1, 2, 3, 4, 5, 6, 7,
                                   8, 9, 10, 11, 12, 13, 14, 15);
}

// ---------------------------------------------------------------------------
// Zero only the padded regions of Q/K (token rows 49..63) and V^T (token
// columns 49..63): ~35 MB of stores instead of a 151 MB memset.
// ---------------------------------------------------------------------------
__global__ __launch_bounds__(256) void zero_pads_kernel(
    _Float16* __restrict__ Qh, _Float16* __restrict__ Kh,
    _Float16* __restrict__ Vt)
{
    int t = blockIdx.x * 256 + threadIdx.x;
    {   // Q/K pads: head in [0,12288), rows 49..63, 32 halves per row
        int d = t & 31, rest = t >> 5;
        int r = rest % 15, head = rest / 15;
        size_t off = (((size_t)head) * NPAD + 49 + r) * HD + d;
        Qh[off] = (_Float16)0.f;
        Kh[off] = (_Float16)0.f;
    }
    {   // V^T pads: row = head*32 + d, cols 49..63
        int c = t % 15, row = t / 15;
        Vt[(size_t)row * NPAD + 49 + c] = (_Float16)0.f;
    }
}

// ---------------------------------------------------------------------------
// Stage 1: fused QKV projection.  784 blocks x 256 threads (8 waves, one
// 16-row M-tile each).  A tiles (x, f32->f16) loaded ONCE into registers;
// Q/K/V weight chunks looped internally; N-tiles processed in 2 halves of 6
// to keep live accumulators at 48 VGPRs (stay under 256-VGPR addressing).
// ---------------------------------------------------------------------------
__global__ __launch_bounds__(256) void qkv_gemm_kernel(
    const float* __restrict__ x, const float* __restrict__ qkv_w,
    const float* __restrict__ qkv_b,
    _Float16* __restrict__ Qh, _Float16* __restrict__ Kh,
    _Float16* __restrict__ Vt)
{
    __shared__ _Float16 wt[DIM * DIM];  // wt[n][k] for current chunk
    const int tid   = threadIdx.x;
    const int wave  = tid >> 5, lane = tid & 31;
    const int mtile = blockIdx.x * 8 + wave;
    const int ncol  = lane & 15;
    const int kb8   = (lane < 16) ? 0 : 8;
    const int kb16  = (lane < 16) ? 0 : 16;
    const int rhalf = (lane < 16) ? 0 : 8;
    const int arow  = mtile * 16 + ncol;

    // A-operand register file: 6 k-tiles of 16x32 (WMMA A layout)
    v16h aq[6];
#pragma unroll
    for (int kt = 0; kt < 6; ++kt) {
        const float* xp = x + (size_t)arow * DIM + kt * 32 + kb8;
#pragma unroll
        for (int i = 0; i < 8; ++i)  aq[kt][i] = (_Float16)xp[i];
#pragma unroll
        for (int i = 8; i < 16; ++i) aq[kt][i] = (_Float16)xp[i + 8];
    }

    for (int chunk = 0; chunk < 3; ++chunk) {
        __syncthreads();  // previous chunk's LDS reads done before refill
        for (int idx = tid; idx < DIM * DIM; idx += 256) {
            int k = idx / DIM, n = idx - k * DIM;
            wt[n * DIM + k] = (_Float16)qkv_w[k * (3 * DIM) + chunk * DIM + n];
        }
        __syncthreads();

        for (int nhalf = 0; nhalf < 2; ++nhalf) {
            v8f acc[6] = {};
#pragma unroll
            for (int kt = 0; kt < 6; ++kt) {
#pragma unroll
                for (int n6 = 0; n6 < 6; ++n6) {
                    int nt = nhalf * 6 + n6;
                    const _Float16* bp = &wt[(nt * 16 + ncol) * DIM + kt * 32 + kb8];
                    v16h bm = merge16(*(const v8h*)bp, *(const v8h*)(bp + 16));
                    acc[n6] = __builtin_amdgcn_wmma_f32_16x16x32_f16(
                        false, aq[kt], false, bm, (short)0, acc[n6],
                        false, false);
                }
            }
            // Epilogue: bias + scatter into attention layouts.
#pragma unroll
            for (int n6 = 0; n6 < 6; ++n6) {
                int c = (nhalf * 6 + n6) * 16 + ncol;
                float bias = qkv_b[chunk * DIM + c];
                int hh = c >> 5, d = c & 31;
#pragma unroll
                for (int j = 0; j < 8; ++j) {
                    int m = mtile * 16 + j + rhalf;
                    int bw = m / NTOK, n = m - bw * NTOK;
                    _Float16 hv = (_Float16)(acc[n6][j] + bias);
                    size_t head = (size_t)bw * H + hh;
                    if (chunk == 0)      Qh[(head * NPAD + n) * HD + d] = hv;
                    else if (chunk == 1) Kh[(head * NPAD + n) * HD + d] = hv;
                    else                 Vt[(head * HD + d) * NPAD + n] = hv;
                }
            }
        }
    }
}

// ---------------------------------------------------------------------------
// Stage 2: attention per window. 1 block = 1 window (2048 blocks), 6 waves =
// 6 heads.  Mask staged via async global->LDS copy (ASYNCcnt), then
// QK^T -> register softmax -> LDS transpose -> attn @ V, all WMMA f16.
// ---------------------------------------------------------------------------
__global__ __launch_bounds__(192) void attn_kernel(
    const _Float16* __restrict__ Qh, const _Float16* __restrict__ Kh,
    const _Float16* __restrict__ Vt, const float* __restrict__ mask,
    _Float16* __restrict__ o16)
{
    __shared__ float    msk[NPAD * NPAD];        // 16 KB
    __shared__ _Float16 attn_s[H][NPAD * NPAD];  // 48 KB
    const int bw  = blockIdx.x;
    const int w   = bw & 63;                     // window index for mask
    const int tid = threadIdx.x;

    // Async-copy the 49x49 mask into the padded 64x64 LDS tile (per-lane
    // scattered GLOBAL_LOAD_ASYNC_TO_LDS_B32); -1e30 padding written to the
    // disjoint pad region by regular DS stores.
    const float* msrc = mask + (size_t)w * NTOK * NTOK;
    for (int idx = tid; idx < NTOK * NTOK; idx += 192) {
        int r = idx / NTOK, c = idx - r * NTOK;
        unsigned lofs = (unsigned)(uintptr_t)&msk[r * NPAD + c];
        const float* g = msrc + idx;
        asm volatile("global_load_async_to_lds_b32 %0, %1, off"
                     :: "v"(lofs), "v"(g) : "memory");
    }
    for (int idx = tid; idx < NPAD * NPAD; idx += 192) {
        int r = idx >> 6, c = idx & 63;
        if (r >= NTOK || c >= NTOK) msk[idx] = -1.0e30f;
    }
    asm volatile("s_wait_asynccnt 0" ::: "memory");
    __syncthreads();

    const int hh = tid >> 5, lane = tid & 31;
    const int ncol  = lane & 15;
    const int kb8   = (lane < 16) ? 0 : 8;
    const int kb16  = (lane < 16) ? 0 : 16;
    const int rhalf = (lane < 16) ? 0 : 8;
    const _Float16* Q = Qh + ((size_t)bw * H + hh) * NPAD * HD;
    const _Float16* K = Kh + ((size_t)bw * H + hh) * NPAD * HD;
    const _Float16* V = Vt + ((size_t)bw * H + hh) * HD * NPAD;

    // --- QK^T: A = Q (16x32 tiles), B = K rows (two 16B runs per lane) ---
    v16h bk[4];
#pragma unroll
    for (int nt = 0; nt < 4; ++nt)
        bk[nt] = merge16(*(const v8h*)&K[(nt * 16 + ncol) * HD + kb8], *(const v8h*)&K[(nt * 16 + ncol) * HD + kb8 + 16]);

    v8f acc[4][4] = {};
#pragma unroll
    for (int mt = 0; mt < 4; ++mt) {
        const _Float16* qp = Q + (mt * 16 + ncol) * HD + kb8;
        v16h a = merge16(*(const v8h*)qp, *(const v8h*)(qp + 16));
#pragma unroll
        for (int nt = 0; nt < 4; ++nt)
            acc[mt][nt] = __builtin_amdgcn_wmma_f32_16x16x32_f16(
                false, a, false, bk[nt], (short)0, acc[mt][nt], false, false);
    }

    // --- softmax over rows (D layout: lane = col, vgpr = row) ---
#pragma unroll
    for (int mt = 0; mt < 4; ++mt) {
#pragma unroll
        for (int j = 0; j < 8; ++j) {
            int r = mt * 16 + j + rhalf;
            float s[4], rmax = -3.0e38f;
#pragma unroll
            for (int nt = 0; nt < 4; ++nt) {
                s[nt] = acc[mt][nt][j] * SCALE + msk[r * NPAD + nt * 16 + ncol];
                rmax = fmaxf(rmax, s[nt]);
            }
#pragma unroll
            for (int off = 1; off < 16; off <<= 1)
                rmax = fmaxf(rmax, __shfl_xor(rmax, off, 32));
            float sum = 0.f;
#pragma unroll
            for (int nt = 0; nt < 4; ++nt) { s[nt] = __expf(s[nt] - rmax); sum += s[nt]; }
#pragma unroll
            for (int off = 1; off < 16; off <<= 1)
                sum += __shfl_xor(sum, off, 32);
            float inv = 1.0f / sum;
#pragma unroll
            for (int nt = 0; nt < 4; ++nt)
                attn_s[hh][r * NPAD + nt * 16 + ncol] = (_Float16)(s[nt] * inv);
        }
    }
    __syncthreads();

    // --- attn @ V: A = probs from LDS, B = V^T rows (contiguous) ---
    const _Float16* As = attn_s[hh];
    v8f oacc[4][2] = {};
#pragma unroll
    for (int ks = 0; ks < 2; ++ks) {
        v16h bv[2];
#pragma unroll
        for (int dt = 0; dt < 2; ++dt)
            bv[dt] = merge16(*(const v8h*)&V[(dt * 16 + ncol) * NPAD + ks * 32 + kb8], *(const v8h*)&V[(dt * 16 + ncol) * NPAD + ks * 32 + kb8 + 16]);
#pragma unroll
        for (int mt = 0; mt < 4; ++mt) {
            const _Float16* ap = As + (mt * 16 + ncol) * NPAD + ks * 32 + kb8;
            v16h a = merge16(*(const v8h*)ap, *(const v8h*)(ap + 16));
#pragma unroll
            for (int dt = 0; dt < 2; ++dt)
                oacc[mt][dt] = __builtin_amdgcn_wmma_f32_16x16x32_f16(
                    false, a, false, bv[dt], (short)0, oacc[mt][dt], false, false);
        }
    }
    // store (b, n, h*32+d) as f16 feeding the projection GEMM
#pragma unroll
    for (int mt = 0; mt < 4; ++mt)
#pragma unroll
        for (int dt = 0; dt < 2; ++dt)
#pragma unroll
            for (int j = 0; j < 8; ++j) {
                int n = mt * 16 + j + rhalf;
                if (n < NTOK) {
                    int d = dt * 16 + ncol;
                    o16[((size_t)bw * NTOK + n) * DIM + hh * HD + d] =
                        (_Float16)oacc[mt][dt][j];
                }
            }
}

// ---------------------------------------------------------------------------
// Stage 3: output projection.  M=100352, K=192, N=192, f32 out + bias.
// ---------------------------------------------------------------------------
__global__ __launch_bounds__(256) void proj_gemm_kernel(
    const _Float16* __restrict__ a16, const float* __restrict__ proj_w,
    const float* __restrict__ proj_b, float* __restrict__ out)
{
    __shared__ _Float16 wt[DIM * DIM];  // wt[n][k]
    const int tid = threadIdx.x;
    for (int idx = tid; idx < DIM * DIM; idx += 256) {
        int k = idx / DIM, n = idx - k * DIM;
        wt[n * DIM + k] = (_Float16)proj_w[idx];
    }
    __syncthreads();

    const int wave  = tid >> 5, lane = tid & 31;
    const int mtile = blockIdx.x * 8 + wave;
    const int ncol  = lane & 15;
    const int kb8   = (lane < 16) ? 0 : 8;
    const int kb16  = (lane < 16) ? 0 : 16;
    const int rhalf = (lane < 16) ? 0 : 8;
    const int arow  = mtile * 16 + ncol;

    v8f acc[12] = {};
    for (int kt = 0; kt < 6; ++kt) {
        const _Float16* ap = a16 + (size_t)arow * DIM + kt * 32 + kb8;
        v16h a = merge16(*(const v8h*)ap, *(const v8h*)(ap + 16));
#pragma unroll
        for (int nt = 0; nt < 12; ++nt) {
            v16h bm = merge16(*(const v8h*)&wt[(nt * 16 + ncol) * DIM + kt * 32 + kb8], *(const v8h*)&wt[(nt * 16 + ncol) * DIM + kt * 32 + kb8 + 16]);
            acc[nt] = __builtin_amdgcn_wmma_f32_16x16x32_f16(
                false, a, false, bm, (short)0, acc[nt], false, false);
        }
    }
#pragma unroll
    for (int nt = 0; nt < 12; ++nt) {
        int c = nt * 16 + ncol;
        float bias = proj_b[c];
#pragma unroll
        for (int j = 0; j < 8; ++j) {
            int m = mtile * 16 + j + rhalf;
            out[(size_t)m * DIM + c] = acc[nt][j] + bias;
        }
    }
}

// ---------------------------------------------------------------------------
extern "C" void kernel_launch(void* const* d_in, const int* in_sizes, int n_in,
                              void* d_out, int out_size, void* d_ws, size_t ws_size,
                              hipStream_t stream)
{
    const float* x      = (const float*)d_in[0];
    const float* mask   = (const float*)d_in[1];
    const float* qkv_w  = (const float*)d_in[2];
    const float* qkv_b  = (const float*)d_in[3];
    const float* proj_w = (const float*)d_in[4];
    const float* proj_b = (const float*)d_in[5];

    char* ws = (char*)d_ws;
    const size_t qbytes = (size_t)BWIN * H * NPAD * HD * sizeof(_Float16); // 50.3 MB
    _Float16* Qh  = (_Float16*)(ws);
    _Float16* Kh  = (_Float16*)(ws + qbytes);
    _Float16* Vt  = (_Float16*)(ws + 2 * qbytes);
    _Float16* o16 = (_Float16*)(ws + 3 * qbytes);

    // zero only the padded token rows/cols (replaces 151 MB memset)
    const int pad_elems = BWIN * H * 15 * 32;  // 5,898,240 for each of Q/K and Vt
    zero_pads_kernel<<<pad_elems / 256, 256, 0, stream>>>(Qh, Kh, Vt);

    qkv_gemm_kernel<<<MROWS / (8 * 16), 256, 0, stream>>>(x, qkv_w, qkv_b,
                                                          Qh, Kh, Vt);
    attn_kernel<<<BWIN, 192, 0, stream>>>(Qh, Kh, Vt, mask, o16);
    proj_gemm_kernel<<<MROWS / (8 * 16), 256, 0, stream>>>(o16, proj_w, proj_b,
                                                           (float*)d_out);
}
